// seq2seq_33681133535392
// MI455X (gfx1250) — matrix operand order repaired and checked by reading, not compiled, not run on hardware
//
#include <hip/hip_runtime.h>
#include <hip/hip_bf16.h>
#include <stdint.h>

// ---------------------------------------------------------------------------
// Problem constants
// ---------------------------------------------------------------------------
#define BB     256      // batch
#define S_IN   1024
#define D_IN   256
#define HH     512
#define G4     2048     // 4*H
#define S_OUT  128
#define D_OUT  8

#define K_ENC0 768      // H + D_IN
#define K_ENC1 1024     // H + H
#define K_DEC0 576      // H + D_OUT + pad(56)  (multiple of 64)
#define K_DEC1 1024

// LDS B-slice: 64 weight rows (4 gates x 16) x 64 halves (one K=64 slice),
// padded to 72 halves (144 B row stride) -> 16-lane column reads hit
// distinct banks (L*36 mod 64 all distinct).
#define BROW   72

typedef __attribute__((ext_vector_type(8)))  _Float16 v8h;
typedef __attribute__((ext_vector_type(16))) _Float16 v16h;
typedef __attribute__((ext_vector_type(8)))  float    v8f;
typedef __attribute__((ext_vector_type(4)))  int      v4i;

// ---------------------------------------------------------------------------
// Device helpers
// ---------------------------------------------------------------------------
__device__ __forceinline__ float sigmoid_f(float x) {
    return 1.0f / (1.0f + __expf(-x));
}

__device__ __forceinline__ v8f wmma_f32_f16(v16h a, v16h b, v8f c) {
#if defined(__HIP_DEVICE_COMPILE__)
    return __builtin_amdgcn_wmma_f32_16x16x32_f16(
        /*neg_a=*/false, a, /*neg_b=*/false, b,
        /*c_mod=*/(short)0, c, /*reuse_a=*/false, /*reuse_b=*/false);
#else
    return c;
#endif
}

// Async DMA: one b128 per lane, global -> LDS, tracked by ASYNCcnt.
__device__ __forceinline__ void async_copy_b128(const _Float16* gsrc,
                                                _Float16* lds_dst) {
#if defined(__HIP_DEVICE_COMPILE__)
# if __has_builtin(__builtin_amdgcn_global_load_async_to_lds_b128)
    __builtin_amdgcn_global_load_async_to_lds_b128(
        (__attribute__((address_space(1))) v4i*)(v4i*)(void*)const_cast<_Float16*>(gsrc),
        (__attribute__((address_space(3))) v4i*)(v4i*)(void*)lds_dst,
        0, 0);
# else
    asm volatile("global_load_async_to_lds_b128 %0, %1, off"
                 :: "v"((unsigned)(unsigned long)lds_dst), "v"(gsrc)
                 : "memory");
# endif
#else
    (void)gsrc; (void)lds_dst;
#endif
}

__device__ __forceinline__ void wait_async0() {
#if defined(__HIP_DEVICE_COMPILE__)
# if __has_builtin(__builtin_amdgcn_s_wait_asynccnt)
    __builtin_amdgcn_s_wait_asynccnt(0);
# else
    asm volatile("s_wait_asynccnt 0x0" ::: "memory");
# endif
#endif
}

// A-matrix 16x32 f16 tile (ISA 7.12.2): lane L holds row m0+(L&15);
// lanes 0-15: K chunks [k0..k0+7] (v0-3) and [k0+16..k0+23] (v4-7);
// lanes 16-31: [k0+8..k0+15] and [k0+24..k0+31].
__device__ __forceinline__ v16h load_a_tile(const _Float16* __restrict__ A,
                                            int lda, int m0, int k0, int lane) {
    int row = m0 + (lane & 15);
    int kb  = k0 + ((lane >> 4) << 3);
    const _Float16* p = A + row * lda + kb;
    v8h lo = *reinterpret_cast<const v8h*>(p);
    v8h hi = *reinterpret_cast<const v8h*>(p + 16);
    return __builtin_shufflevector(lo, hi, 0,1,2,3,4,5,6,7,8,9,10,11,12,13,14,15);
}

// B-matrix 32x16 f16 tile out of the LDS-staged K=64 slice (7.12.4 B layout):
// lane L holds column row (gate*16 + (L&15)); lanes 0-15 hold K=+0..7,+16..23,
// lanes 16-31 hold K=+8..15,+24..31 of half-slice `half` (0 or 1).
__device__ __forceinline__ v16h lds_b_tile(const _Float16* buf, int gate,
                                           int half, int lane) {
    const _Float16* p = buf + (gate * 16 + (lane & 15)) * BROW
                            + half * 32 + ((lane >> 4) << 3);
    v8h lo = *reinterpret_cast<const v8h*>(p);
    v8h hi = *reinterpret_cast<const v8h*>(p + 16);
    return __builtin_shufflevector(lo, hi, 0,1,2,3,4,5,6,7,8,9,10,11,12,13,14,15);
}

// Cooperative async stage of one 64-wide k-slice of B for all 4 gates:
// 64 rows x 128 B; 128 threads issue 4 b128 each.
__device__ __forceinline__ void stage_b_slice(const _Float16* __restrict__ W,
                                              int ldw, int n0, int k0,
                                              _Float16* buf, int tid) {
    int r = tid >> 1;                               // 0..63 (gate*16 + row)
    int cb = (tid & 1) * 32;                        // half-row (64 B) base
    int wrow = n0 + ((r >> 4) << 9) + (r & 15);     // gate stride 512
    const _Float16* g = W + wrow * ldw + k0 + cb;
    _Float16* l = buf + r * BROW + cb;
#pragma unroll
    for (int c = 0; c < 4; ++c)
        async_copy_b128(g + c * 8, l + c * 8);
}

// ---------------------------------------------------------------------------
// Fused LSTM step: gates = A @ W^T (+bias), cell update, h -> fp16 dests.
// A: [256 x K] fp16 row-major (cols 0..511 = h_prev, rest = step input)
// W: [2048 x K] fp16 row-major, PyTorch gate row order i,f,g,o (chunks of 512)
// Each wave computes TWO 16(M)x16(N) tiles for all 4 gates (N-stride 512):
// the wave owns complete (i,f,g,o) quadruples -> cell update fused in
// registers. B k-slices (K=64) are async-DMA'd to LDS, double buffered,
// shared by the 4 waves. Grid: 64 blocks x 128 threads
// (2 M-halves x 32 N-tiles; block = 4 waves x 2 M-tiles).
// ---------------------------------------------------------------------------
__global__ __launch_bounds__(128) void lstm_step_kernel(
    const _Float16* __restrict__ A, int lda, int K,
    const _Float16* __restrict__ W, int ldw,
    const float* __restrict__ bias,
    float* __restrict__ cstate,
    _Float16* __restrict__ hdst0, int ld0, int off0,
    _Float16* __restrict__ hdst1, int ld1, int off1,
    const float* __restrict__ xsrc, long xsrc_stride,
    _Float16* __restrict__ xdst, int xd_ld, int xd_off, int xd_cols)
{
    __shared__ _Float16 bufB[2][64 * BROW];

    const int tid  = threadIdx.x;
    const int lane = tid & 31;
    const int wave = tid >> 5;
    const int m0a = ((blockIdx.x & 1) * 4 + wave) * 32;  // first M tile
    const int m0b = m0a + 16;                            // second M tile
    const int n0  = (blockIdx.x >> 1) * 16;              // 32 N-base tiles

    v8f ai0 = {}, af0 = {}, ag0 = {}, ao0 = {};   // accum for m0a
    v8f ai1 = {}, af1 = {}, ag1 = {}, ao1 = {};   // accum for m0b

    // Prologue: DMA first k-slice of B into buffer 0.
    stage_b_slice(W, ldw, n0, 0, bufB[0], tid);

    int cur = 0;
    for (int k0 = 0; k0 < K; k0 += 64) {
        wait_async0();           // our slice writes have landed in LDS
        __syncthreads();         // everyone's writes visible; prev reads done
        if (k0 + 64 < K)         // overlap next slice DMA with this slice's math
            stage_b_slice(W, ldw, n0, k0 + 64, bufB[cur ^ 1], tid);

#pragma unroll
        for (int h = 0; h < 2; ++h) {
            const int kh = k0 + h * 32;
            // hoist ALL loads of this half-slice before any WMMA so a single
            // counter wait covers them and the 8 WMMAs issue back-to-back
            v16h a0 = load_a_tile(A, lda, m0a, kh, lane);
            v16h a1 = load_a_tile(A, lda, m0b, kh, lane);
            v16h b0 = lds_b_tile(bufB[cur], 0, h, lane);
            v16h b1 = lds_b_tile(bufB[cur], 1, h, lane);
            v16h b2 = lds_b_tile(bufB[cur], 2, h, lane);
            v16h b3 = lds_b_tile(bufB[cur], 3, h, lane);
            ai0 = wmma_f32_f16(a0, b0, ai0);
            ai1 = wmma_f32_f16(a1, b0, ai1);
            af0 = wmma_f32_f16(a0, b1, af0);
            af1 = wmma_f32_f16(a1, b1, af1);
            ag0 = wmma_f32_f16(a0, b2, ag0);
            ag1 = wmma_f32_f16(a1, b2, ag1);
            ao0 = wmma_f32_f16(a0, b3, ao0);
            ao1 = wmma_f32_f16(a1, b3, ao1);
        }
        cur ^= 1;
    }

    // Epilogue: C/D layout -> lane owns column j = n0+(lane&15);
    // VGPR vg holds row m0 + vg + (lane>=16 ? 8 : 0).
    const int j  = n0 + (lane & 15);
    const float b_i = bias[j];
    const float b_f = bias[512  + j];
    const float b_g = bias[1024 + j];
    const float b_o = bias[1536 + j];
    const int hi8 = (lane >> 4) << 3;

#pragma unroll
    for (int mt = 0; mt < 2; ++mt) {
        const int mbase = (mt ? m0b : m0a) + hi8;
        v8f acc_i = mt ? ai1 : ai0;
        v8f acc_f = mt ? af1 : af0;
        v8f acc_g = mt ? ag1 : ag0;
        v8f acc_o = mt ? ao1 : ao0;
#pragma unroll
        for (int vg = 0; vg < 8; ++vg) {
            const int m = mbase + vg;
            float gi = acc_i[vg] + b_i;
            float gf = acc_f[vg] + b_f;
            float gg = acc_g[vg] + b_g;
            float go = acc_o[vg] + b_o;
            float c_old = cstate[m * HH + j];
            float c_new = sigmoid_f(gf) * c_old + sigmoid_f(gi) * tanhf(gg);
            cstate[m * HH + j] = c_new;
            float h = sigmoid_f(go) * tanhf(c_new);
            _Float16 h16 = (_Float16)h;
            hdst0[m * ld0 + off0 + j] = h16;
            if (hdst1) hdst1[m * ld1 + off1 + j] = h16;
        }
    }

    // Optional: stage next-step x (fp32 -> fp16) into the next A buffer.
    if (xdst) {
        const int total = BB * xd_cols;
        const int stride = gridDim.x * blockDim.x;
        for (int i = blockIdx.x * blockDim.x + tid; i < total; i += stride) {
            int b = i / xd_cols, d = i - b * xd_cols;
            xdst[b * xd_ld + xd_off + d] = (_Float16)xsrc[(long)b * xsrc_stride + d];
        }
    }
}

// ---------------------------------------------------------------------------
// Helper kernels
// ---------------------------------------------------------------------------
__global__ void zero_u32_kernel(uint32_t* p, int n) {
    for (int i = blockIdx.x * blockDim.x + threadIdx.x; i < n;
         i += gridDim.x * blockDim.x) p[i] = 0u;
}

// dst[g*ldd + k] = k<ka ? a[g*ka+k] : (k<ka+kb ? b[g*kb+(k-ka)] : 0)   fp32->fp16
__global__ void pack_w_kernel(_Float16* dst, int ldd,
                              const float* a, int ka,
                              const float* b, int kb) {
    const int total = G4 * ldd;
    for (int i = blockIdx.x * blockDim.x + threadIdx.x; i < total;
         i += gridDim.x * blockDim.x) {
        int g = i / ldd, k = i - g * ldd;
        float v = 0.0f;
        if (k < ka)            v = a[g * ka + k];
        else if (k < ka + kb)  v = b[g * kb + (k - ka)];
        dst[i] = (_Float16)v;
    }
}

// fp32 [rows x cols] (row stride srs) -> fp16 dst[r*dld + doff + c]
__global__ void stage_f32_f16_kernel(_Float16* dst, int dld, int doff,
                                     const float* src, long srs,
                                     int rows, int cols) {
    const int total = rows * cols;
    for (int i = blockIdx.x * blockDim.x + threadIdx.x; i < total;
         i += gridDim.x * blockDim.x) {
        int r = i / cols, c = i - r * cols;
        dst[r * dld + doff + c] = (_Float16)src[(long)r * srs + c];
    }
}

// fp16 2D copy
__global__ void copy_h16_kernel(_Float16* dst, int dld,
                                const _Float16* src, int sld,
                                int rows, int cols) {
    const int total = rows * cols;
    for (int i = blockIdx.x * blockDim.x + threadIdx.x; i < total;
         i += gridDim.x * blockDim.x) {
        int r = i / cols, c = i - r * cols;
        dst[r * dld + c] = src[r * sld + c];
    }
}

// y[b,j] = h1[b,:] . fcW[j,:] + fcb[j]; write fp32 out and fp16 next-input.
__global__ __launch_bounds__(256) void fc_kernel(
    const _Float16* __restrict__ h, int hld,
    const float* __restrict__ fcW, const float* __restrict__ fcb,
    float* __restrict__ out, int out_row_stride,
    _Float16* __restrict__ ydst, int yld, int yoff)
{
    const int tid = blockIdx.x * 256 + threadIdx.x;     // 2048 threads
    if (tid >= BB * D_OUT) return;
    const int b = tid >> 3;
    const int j = tid & 7;
    float s = fcb[j];
    const _Float16* hp = h + b * hld;
    const float*    wp = fcW + j * HH;
#pragma unroll 4
    for (int k = 0; k < HH; ++k) s += (float)hp[k] * wp[k];
    out[b * out_row_stride + j] = s;
    ydst[b * yld + yoff + j] = (_Float16)s;
}

// ---------------------------------------------------------------------------
// Host launch
// ---------------------------------------------------------------------------
extern "C" void kernel_launch(void* const* d_in, const int* in_sizes, int n_in,
                              void* d_out, int out_size, void* d_ws, size_t ws_size,
                              hipStream_t stream) {
    (void)in_sizes; (void)n_in; (void)out_size; (void)ws_size;

    const float* x        = (const float*)d_in[0];
    const float* enc_Wih0 = (const float*)d_in[2];
    const float* enc_Whh0 = (const float*)d_in[3];
    const float* enc_b0   = (const float*)d_in[4];
    const float* enc_Wih1 = (const float*)d_in[5];
    const float* enc_Whh1 = (const float*)d_in[6];
    const float* enc_b1   = (const float*)d_in[7];
    const float* dec_Wih0 = (const float*)d_in[8];
    const float* dec_Whh0 = (const float*)d_in[9];
    const float* dec_b0   = (const float*)d_in[10];
    const float* dec_Wih1 = (const float*)d_in[11];
    const float* dec_Whh1 = (const float*)d_in[12];
    const float* dec_b1   = (const float*)d_in[13];
    const float* fc_W     = (const float*)d_in[14];
    const float* fc_b     = (const float*)d_in[15];
    float* out = (float*)d_out;

    // ---- workspace layout ----
    char* ws = (char*)d_ws;
    size_t off = 0;
    auto alloc = [&](size_t bytes) -> char* {
        char* p = ws + off;
        off = (off + bytes + 255) & ~(size_t)255;
        return p;
    };
    _Float16* W0p  = (_Float16*)alloc((size_t)G4 * K_ENC0 * 2);
    _Float16* W1p  = (_Float16*)alloc((size_t)G4 * K_ENC1 * 2);
    _Float16* W0dp = (_Float16*)alloc((size_t)G4 * K_DEC0 * 2);
    _Float16* W1dp = (_Float16*)alloc((size_t)G4 * K_DEC1 * 2);
    size_t state_begin = off;
    _Float16* A0[2]  = {(_Float16*)alloc((size_t)BB * K_ENC0 * 2),
                        (_Float16*)alloc((size_t)BB * K_ENC0 * 2)};
    _Float16* A1[2]  = {(_Float16*)alloc((size_t)BB * K_ENC1 * 2),
                        (_Float16*)alloc((size_t)BB * K_ENC1 * 2)};
    _Float16* A0d[2] = {(_Float16*)alloc((size_t)BB * K_DEC0 * 2),
                        (_Float16*)alloc((size_t)BB * K_DEC0 * 2)};
    _Float16* A1d[2] = {(_Float16*)alloc((size_t)BB * K_DEC1 * 2),
                        (_Float16*)alloc((size_t)BB * K_DEC1 * 2)};
    float* c0 = (float*)alloc((size_t)BB * HH * 4);
    float* c1 = (float*)alloc((size_t)BB * HH * 4);
    size_t state_end = off;

    // ---- init: zero all recurrent state (h buffers, pads, c) ----
    {
        int nwords = (int)((state_end - state_begin) / 4);
        zero_u32_kernel<<<512, 256, 0, stream>>>((uint32_t*)(ws + state_begin), nwords);
    }
    // ---- pack weights fp32 -> fp16, [Whh | Wih | pad] ----
    pack_w_kernel<<<1024, 256, 0, stream>>>(W0p,  K_ENC0, enc_Whh0, HH, enc_Wih0, D_IN);
    pack_w_kernel<<<1024, 256, 0, stream>>>(W1p,  K_ENC1, enc_Whh1, HH, enc_Wih1, HH);
    pack_w_kernel<<<1024, 256, 0, stream>>>(W0dp, K_DEC0, dec_Whh0, HH, dec_Wih0, D_OUT);
    pack_w_kernel<<<1024, 256, 0, stream>>>(W1dp, K_DEC1, dec_Whh1, HH, dec_Wih1, HH);
    // ---- stage x_0 into A0[0] cols 512..767 ----
    stage_f32_f16_kernel<<<256, 256, 0, stream>>>(A0[0], K_ENC0, HH,
                                                  x, (long)S_IN * D_IN, BB, D_IN);

    // ---- encoder: 1024 steps, 2 layers ----
    int p = 0;
    for (int t = 0; t < S_IN; ++t) {
        // layer 0: reads A0[p] ([h0(t-1)|x_t]); h0(t) -> A0[1-p] and A1[p] cols512+
        lstm_step_kernel<<<64, 128, 0, stream>>>(
            A0[p], K_ENC0, K_ENC0, W0p, K_ENC0, enc_b0, c0,
            A0[1 - p], K_ENC0, 0,
            A1[p],     K_ENC1, HH,
            nullptr, 0, nullptr, 0, 0, 0);
        // layer 1: reads A1[p] ([h1(t-1)|h0(t)]); h1(t) -> A1[1-p];
        // epilogue stages x_{t+1} into A0[1-p].
        const float* xs = (t + 1 < S_IN) ? (x + (size_t)(t + 1) * D_IN) : nullptr;
        lstm_step_kernel<<<64, 128, 0, stream>>>(
            A1[p], K_ENC1, K_ENC1, W1p, K_ENC1, enc_b1, c1,
            A1[1 - p], K_ENC1, 0,
            nullptr, 0, 0,
            xs, (long)S_IN * D_IN,
            xs ? A0[1 - p] : nullptr, K_ENC0, HH, D_IN);
        p ^= 1;
    }
    // after loop p == 0: final h0 in A0[0], final h1 in A1[0]

    // ---- hand state to decoder ----
    copy_h16_kernel<<<256, 256, 0, stream>>>(A0d[0], K_DEC0, A0[0], K_ENC0, BB, HH);
    copy_h16_kernel<<<256, 256, 0, stream>>>(A1d[0], K_DEC1, A1[0], K_ENC1, BB, HH);
    // c0/c1 carry over in place; A0d[0] input cols (512..575) are zero (=inp0).

    // ---- decoder: 128 steps, 2 layers + fc ----
    int q = 0;
    for (int t = 0; t < S_OUT; ++t) {
        lstm_step_kernel<<<64, 128, 0, stream>>>(
            A0d[q], K_DEC0, K_DEC0, W0dp, K_DEC0, dec_b0, c0,
            A0d[1 - q], K_DEC0, 0,
            A1d[q],     K_DEC1, HH,
            nullptr, 0, nullptr, 0, 0, 0);
        lstm_step_kernel<<<64, 128, 0, stream>>>(
            A1d[q], K_DEC1, K_DEC1, W1dp, K_DEC1, dec_b1, c1,
            A1d[1 - q], K_DEC1, 0,
            nullptr, 0, 0,
            nullptr, 0, nullptr, 0, 0, 0);
        // y_t = h1 @ fcW^T + fcb ; also feeds next decoder input.
        fc_kernel<<<8, 256, 0, stream>>>(
            A1d[1 - q], K_DEC1, fc_W, fc_b,
            out + (size_t)t * D_OUT, S_OUT * D_OUT,
            A0d[1 - q], K_DEC0, HH);
        q ^= 1;
    }
}
